// TransitionDown_32538672234530
// MI455X (gfx1250) — matrix operand order repaired and checked by reading, not compiled, not run on hardware
//
#include <hip/hip_runtime.h>
#include <hip/hip_bf16.h>

#define Bn 8
#define Nn 8192
#define Sn 2048
#define Kn 16
#define DIN 128
#define DOUT 256
#define Mrows (Bn * Sn * Kn)   // 262144
#define EPSV 1e-5f
#define NTW 8                  // N-tiles (of 16 channels) per wave

typedef __attribute__((ext_vector_type(16))) __bf16 v16bf;
typedef __attribute__((ext_vector_type(8)))  __bf16 v8bf;
typedef __attribute__((ext_vector_type(8)))  float  v8f;

// ---------------------------------------------------------------------------
// util kernels
// ---------------------------------------------------------------------------
__global__ void zero_f32_kernel(float* p, int n) {
    int i = blockIdx.x * blockDim.x + threadIdx.x;
    if (i < n) p[i] = 0.f;
}

__global__ void f32_to_bf16_kernel(const float* __restrict__ in, __bf16* __restrict__ out, int n) {
    int i = blockIdx.x * blockDim.x + threadIdx.x;
    if (i < n) out[i] = (__bf16)in[i];
}

// ---------------------------------------------------------------------------
// Farthest point sampling: one block (1024 threads) per batch.
// Points + running min-distance live in registers; block argmax each step.
// ---------------------------------------------------------------------------
__global__ __launch_bounds__(1024) void fps_kernel(const float* __restrict__ xyz,
                                                   int* __restrict__ fps_idx,
                                                   float* __restrict__ out_xyz) {
    const int b = blockIdx.x;
    const int t = threadIdx.x;
    const float* xb = xyz + (size_t)b * Nn * 3;

    float px[8], py[8], pz[8], dist[8];
#pragma unroll
    for (int j = 0; j < 8; ++j) {
        int p = t + j * 1024;
        px[j] = xb[p * 3 + 0];
        py[j] = xb[p * 3 + 1];
        pz[j] = xb[p * 3 + 2];
        dist[j] = 1e10f;
    }

    __shared__ unsigned long long warr[32];
    __shared__ int s_far;
    const int lane = t & 31, wid = t >> 5;

    int far = 0;  // reference seeds randomly; deterministic seed 0 here
    for (int s = 0; s < Sn; ++s) {
        if (t == 0) {
            fps_idx[b * Sn + s] = far;
            size_t o = ((size_t)b * Sn + s) * 3;
            out_xyz[o + 0] = xb[far * 3 + 0];
            out_xyz[o + 1] = xb[far * 3 + 1];
            out_xyz[o + 2] = xb[far * 3 + 2];
        }
        const float cx = xb[far * 3 + 0], cy = xb[far * 3 + 1], cz = xb[far * 3 + 2];

        float bestd = -1.f;
        int bestp = 0;
#pragma unroll
        for (int j = 0; j < 8; ++j) {
            float dx = px[j] - cx, dy = py[j] - cy, dz = pz[j] - cz;
            float d = dx * dx + dy * dy + dz * dz;
            dist[j] = fminf(dist[j], d);
            if (dist[j] > bestd) { bestd = dist[j]; bestp = t + j * 1024; }
        }
        unsigned long long pk =
            ((unsigned long long)__float_as_uint(bestd) << 32) | (unsigned)bestp;
#pragma unroll
        for (int off = 16; off > 0; off >>= 1) {
            unsigned long long o = __shfl_xor(pk, off, 32);
            pk = (o > pk) ? o : pk;
        }
        if (lane == 0) warr[wid] = pk;
        __syncthreads();
        if (wid == 0) {
            unsigned long long v = warr[lane];
#pragma unroll
            for (int off = 16; off > 0; off >>= 1) {
                unsigned long long o = __shfl_xor(v, off, 32);
                v = (o > v) ? o : v;
            }
            if (lane == 0) s_far = (int)(v & 0xffffffffull);
        }
        __syncthreads();
        far = s_far;
    }
}

// ---------------------------------------------------------------------------
// kNN (K=16 nearest of N=8192): one wave per query; per-lane register top-16,
// then a 512-candidate LDS selection of the 16 global minima.
// ---------------------------------------------------------------------------
__global__ __launch_bounds__(256) void knn_kernel(const float* __restrict__ xyz,
                                                  const float* __restrict__ new_xyz,
                                                  int* __restrict__ knn_idx) {
    __shared__ float sd[8][512];
    __shared__ int   si[8][512];
    const int w = threadIdx.x >> 5, lane = threadIdx.x & 31;
    const int q = blockIdx.x * 8 + w;      // 0 .. B*S-1
    const int b = q >> 11;                 // / Sn
    const float* xb = xyz + (size_t)b * Nn * 3;
    const float qx = new_xyz[q * 3 + 0];
    const float qy = new_xyz[q * 3 + 1];
    const float qz = new_xyz[q * 3 + 2];

    float bd[16]; int bi[16];
#pragma unroll
    for (int i = 0; i < 16; ++i) { bd[i] = 1e30f; bi[i] = 0; }
    float bmax = 1e30f; int bslot = 0;

    for (int j = 0; j < 256; ++j) {
        int p = lane + (j << 5);
        float dx = xb[p * 3 + 0] - qx;
        float dy = xb[p * 3 + 1] - qy;
        float dz = xb[p * 3 + 2] - qz;
        float d = dx * dx + dy * dy + dz * dz;
        if (d < bmax) {
#pragma unroll
            for (int i = 0; i < 16; ++i)
                if (i == bslot) { bd[i] = d; bi[i] = p; }
            bmax = -1.f;
#pragma unroll
            for (int i = 0; i < 16; ++i)
                if (bd[i] > bmax) { bmax = bd[i]; bslot = i; }
        }
    }
#pragma unroll
    for (int i = 0; i < 16; ++i) {
        sd[w][lane * 16 + i] = bd[i];
        si[w][lane * 16 + i] = bi[i];
    }
    __syncthreads();

    for (int r = 0; r < Kn; ++r) {
        float mv = 1e31f; int me = lane * 16;
#pragma unroll
        for (int i = 0; i < 16; ++i) {
            float v = sd[w][lane * 16 + i];
            if (v < mv) { mv = v; me = lane * 16 + i; }
        }
        unsigned long long pk =
            ((unsigned long long)__float_as_uint(mv) << 32) | (unsigned)me;
#pragma unroll
        for (int off = 16; off > 0; off >>= 1) {
            unsigned long long o = __shfl_xor(pk, off, 32);
            pk = (o < pk) ? o : pk;
        }
        int e = (int)(pk & 0xffffffffull);
        if (lane == 0) {
            knn_idx[(size_t)q * Kn + r] = si[w][e];
            sd[w][e] = 1e30f;
        }
        __syncthreads();
    }
}

// ---------------------------------------------------------------------------
// Gather features[b, idx] -> A1 (bf16, M x 128 row-major)
// ---------------------------------------------------------------------------
__global__ __launch_bounds__(256) void gather_kernel(const float* __restrict__ feats,
                                                     const int* __restrict__ knn_idx,
                                                     __bf16* __restrict__ A1) {
    size_t u = (size_t)blockIdx.x * 256 + threadIdx.x;  // M*16 threads
    size_t m = u >> 4;
    int c8 = (int)(u & 15);
    int b = (int)(m >> 15);                 // / (Sn*Kn)
    int i = knn_idx[m];
    const float* src = feats + ((size_t)b * Nn + i) * DIN + c8 * 8;
    float4 f0 = *(const float4*)(src);
    float4 f1 = *(const float4*)(src + 4);
    v8bf o;
    o[0] = (__bf16)f0.x; o[1] = (__bf16)f0.y; o[2] = (__bf16)f0.z; o[3] = (__bf16)f0.w;
    o[4] = (__bf16)f1.x; o[5] = (__bf16)f1.y; o[6] = (__bf16)f1.z; o[7] = (__bf16)f1.w;
    *(v8bf*)(A1 + m * DIN + c8 * 8) = o;
}

// ---------------------------------------------------------------------------
// Register-blocked WMMA GEMM: H[M x 256] = A[M x Kdim] * W^T
// (W stored [256 x Kdim] row-major, bf16). One wave owns one 16-row M-tile and
// NTW=8 N-tiles (128 channels): the A fragment is loaded once per k-step and
// replayed against 8 cached weight fragments -> 8x less activation traffic.
// Fused per-channel sum/sumsq for training-mode BN.
// grid = (Mtiles/8, 2), block = 256 (8 waves).
// ---------------------------------------------------------------------------
__global__ __launch_bounds__(256) void gemm_bf16_kernel(const __bf16* __restrict__ A,
                                                        const __bf16* __restrict__ Bw,
                                                        __bf16* __restrict__ H,
                                                        float* __restrict__ gsum,
                                                        float* __restrict__ gsq,
                                                        int Kdim) {
    __shared__ float lsum[NTW * 16], lsq[NTW * 16];
    const int w = threadIdx.x >> 5, lane = threadIdx.x & 31;
    for (int i = threadIdx.x; i < NTW * 16; i += 256) { lsum[i] = 0.f; lsq[i] = 0.f; }
    __syncthreads();

    const int mt = blockIdx.x * 8 + w;
    const int nbase = blockIdx.y * (NTW * 16);   // 0 or 128
    const int nl = lane & 15, kh = lane >> 4;
    const int m = mt * 16 + nl;                  // A row of this lane

    const __bf16* arow = A + (size_t)m * Kdim + kh * 16;             // 16-half chunk as two 8-half loads
    const __bf16* brow = Bw + (size_t)(nbase + nl) * Kdim + kh * 16; // 16-half chunk, +j*16*Kdim

    v8f acc[NTW];
#pragma unroll
    for (int j = 0; j < NTW; ++j) acc[j] = (v8f){};

    const int kiters = Kdim >> 5;
    for (int kk = 0; kk < kiters; ++kk) {
        __builtin_prefetch(arow + (kk + 1) * 32, 0, 0);   // global_prefetch_b8
        v8bf a0 = *(const v8bf*)(arow + kk * 32);
        v8bf a1 = *(const v8bf*)(arow + kk * 32 + 8);
        v16bf av;
#pragma unroll
        for (int i = 0; i < 8; ++i) { av[i] = a0[i]; av[i + 8] = a1[i]; }
#pragma unroll
        for (int j = 0; j < NTW; ++j) {
            v16bf bv = *(const v16bf*)(brow + (size_t)j * 16 * Kdim + kk * 32);
            acc[j] = __builtin_amdgcn_wmma_f32_16x16x32_bf16(
                false, av, false, bv, (short)0, acc[j], false, false);
        }
    }

    // D store + per-channel statistics (lane L and L^16 share channel)
    size_t hbase = ((size_t)mt * 16 + 8 * kh) * DOUT + nbase + nl;
#pragma unroll
    for (int j = 0; j < NTW; ++j) {
        float s = 0.f, qa = 0.f;
#pragma unroll
        for (int r = 0; r < 8; ++r) {
            float v = acc[j][r];
            s += v; qa += v * v;
            H[hbase + (size_t)r * DOUT + j * 16] = (__bf16)v;
        }
        s  += __shfl_xor(s, 16, 32);
        qa += __shfl_xor(qa, 16, 32);
        if (lane < 16) {
            atomicAdd(&lsum[j * 16 + lane], s);
            atomicAdd(&lsq[j * 16 + lane], qa);
        }
    }
    __syncthreads();
    for (int i = threadIdx.x; i < NTW * 16; i += 256) {
        atomicAdd(&gsum[nbase + i], lsum[i]);
        atomicAdd(&gsq[nbase + i], lsq[i]);
    }
}

// ---------------------------------------------------------------------------
// BN finalize: scale = gamma*rsqrt(var+eps), shift = beta - mean*scale.
// (conv bias cancels exactly under training-mode BN, so it is omitted)
// ---------------------------------------------------------------------------
__global__ void bn_finalize_kernel(const float* gsum, const float* gsq,
                                   const float* gamma, const float* beta,
                                   float* scale, float* shift, float invM) {
    int n = threadIdx.x;
    float mean = gsum[n] * invM;
    float var = gsq[n] * invM - mean * mean;
    float sc = gamma[n] * rsqrtf(var + EPSV);
    scale[n] = sc;
    shift[n] = beta[n] - mean * sc;
}

// BN + ReLU applied in place on bf16 activations (prep for GEMM2 A-matrix)
__global__ __launch_bounds__(256) void bnrelu_kernel(__bf16* __restrict__ h,
                                                     const float* __restrict__ scale,
                                                     const float* __restrict__ shift) {
    size_t u = (size_t)blockIdx.x * 256 + threadIdx.x;  // M*256/8 threads
    int ch = (int)((u & 31) << 3);
    v8bf v = *(v8bf*)(h + u * 8);
    v8bf o;
#pragma unroll
    for (int i = 0; i < 8; ++i) {
        float f = (float)v[i] * scale[ch + i] + shift[ch + i];
        o[i] = (__bf16)fmaxf(f, 0.f);
    }
    *(v8bf*)(h + u * 8) = o;
}

// BN + ReLU + max over K, writes new_features (fp32) into d_out
__global__ __launch_bounds__(256) void maxpool_kernel(const __bf16* __restrict__ h2,
                                                      const float* __restrict__ scale,
                                                      const float* __restrict__ shift,
                                                      float* __restrict__ out) {
    int u = blockIdx.x * 256 + threadIdx.x;  // B*S*32 threads
    int bs = u >> 5;
    int ch = (u & 31) << 3;
    float sc[8], sh[8], mx[8];
#pragma unroll
    for (int i = 0; i < 8; ++i) {
        sc[i] = scale[ch + i];
        sh[i] = shift[ch + i];
        mx[i] = -1e30f;
    }
    for (int k = 0; k < Kn; ++k) {
        v8bf v = *(const v8bf*)(h2 + ((size_t)(bs * Kn + k)) * DOUT + ch);
#pragma unroll
        for (int i = 0; i < 8; ++i)
            mx[i] = fmaxf(mx[i], (float)v[i] * sc[i] + sh[i]);
    }
    float* dst = out + (size_t)Bn * Sn * 3 + (size_t)bs * DOUT + ch;
#pragma unroll
    for (int i = 0; i < 8; ++i) dst[i] = fmaxf(mx[i], 0.f);
}

// ---------------------------------------------------------------------------
extern "C" void kernel_launch(void* const* d_in, const int* in_sizes, int n_in,
                              void* d_out, int out_size, void* d_ws, size_t ws_size,
                              hipStream_t stream) {
    const float* xyz   = (const float*)d_in[0];
    const float* feats = (const float*)d_in[1];
    const float* W1    = (const float*)d_in[2];
    // d_in[3] = b1 (cancels under training BN)
    const float* g1    = (const float*)d_in[4];
    const float* be1   = (const float*)d_in[5];
    const float* W2    = (const float*)d_in[6];
    // d_in[7] = b2 (cancels under training BN)
    const float* g2    = (const float*)d_in[8];
    const float* be2   = (const float*)d_in[9];
    float* out = (float*)d_out;

    // workspace carve-up (4KB aligned)
    char* ws = (char*)d_ws;
    size_t off = 0;
    auto carve = [&](size_t bytes) {
        char* p = ws + off;
        off = (off + bytes + 4095) & ~(size_t)4095;
        return p;
    };
    int*    fps_idx = (int*)carve((size_t)Bn * Sn * 4);
    int*    knn_idx = (int*)carve((size_t)Mrows * 4);
    float*  sum1    = (float*)carve(256 * 4);
    float*  sq1     = (float*)carve(256 * 4);
    float*  sum2    = (float*)carve(256 * 4);
    float*  sq2     = (float*)carve(256 * 4);
    float*  scale1  = (float*)carve(256 * 4);
    float*  shift1  = (float*)carve(256 * 4);
    float*  scale2  = (float*)carve(256 * 4);
    float*  shift2  = (float*)carve(256 * 4);
    __bf16* W1b     = (__bf16*)carve((size_t)DOUT * DIN * 2);
    __bf16* W2b     = (__bf16*)carve((size_t)DOUT * DOUT * 2);
    __bf16* A1      = (__bf16*)carve((size_t)Mrows * DIN * 2);
    __bf16* H1      = (__bf16*)carve((size_t)Mrows * DOUT * 2);
    __bf16* H2      = (__bf16*)carve((size_t)Mrows * DOUT * 2);

    const float invM = 1.0f / (float)Mrows;

    // stats accumulators must start at zero (ws is poisoned)
    zero_f32_kernel<<<1, 256, 0, stream>>>(sum1, 256);
    zero_f32_kernel<<<1, 256, 0, stream>>>(sq1, 256);
    zero_f32_kernel<<<1, 256, 0, stream>>>(sum2, 256);
    zero_f32_kernel<<<1, 256, 0, stream>>>(sq2, 256);

    // weights -> bf16
    f32_to_bf16_kernel<<<(DOUT * DIN) / 256, 256, 0, stream>>>(W1, W1b, DOUT * DIN);
    f32_to_bf16_kernel<<<(DOUT * DOUT) / 256, 256, 0, stream>>>(W2, W2b, DOUT * DOUT);

    // 1) FPS -> fps_idx, new_xyz (front of d_out)
    fps_kernel<<<Bn, 1024, 0, stream>>>(xyz, fps_idx, out);

    // 2) kNN -> knn_idx
    knn_kernel<<<(Bn * Sn) / 8, 256, 0, stream>>>(xyz, out, knn_idx);

    // 3) gather -> A1 bf16
    gather_kernel<<<(Mrows * 16) / 256, 256, 0, stream>>>(feats, knn_idx, A1);

    // 4) GEMM1 (M x 128 x 256) + fused BN stats
    gemm_bf16_kernel<<<dim3((Mrows / 16) / 8, 2), 256, 0, stream>>>(
        A1, W1b, H1, sum1, sq1, DIN);
    bn_finalize_kernel<<<1, 256, 0, stream>>>(sum1, sq1, g1, be1, scale1, shift1, invM);
    bnrelu_kernel<<<(Mrows * DOUT / 8) / 256, 256, 0, stream>>>(H1, scale1, shift1);

    // 5) GEMM2 (M x 256 x 256) + fused BN stats
    gemm_bf16_kernel<<<dim3((Mrows / 16) / 8, 2), 256, 0, stream>>>(
        H1, W2b, H2, sum2, sq2, DOUT);
    bn_finalize_kernel<<<1, 256, 0, stream>>>(sum2, sq2, g2, be2, scale2, shift2, invM);

    // 6) BN + ReLU + max over K -> new_features in d_out
    maxpool_kernel<<<(Bn * Sn * 32) / 256, 256, 0, stream>>>(H2, scale2, shift2, out);
}
